// DestationaryAttention_5798205849703
// MI455X (gfx1250) — matrix operand order repaired and checked by reading, not compiled, not run on hardware
//
#include <hip/hip_runtime.h>

// ---------------------------------------------------------------------------
// DestationaryAttention for MI455X (gfx1250, wave32, WMMA bf16 16x16x32)
// B=4, T=2048, D=1024, H=16, HD=64
// Async LDS pipeline: GLOBAL_LOAD_ASYNC_TO_LDS_B128 + s_wait_asynccnt
// ---------------------------------------------------------------------------

#define Bn 4
#define Tn 2048
#define Dn 1024
#define Hn 16
#define HDn 64
#define NTOK (Bn * Tn)          // 8192 tokens

typedef __attribute__((ext_vector_type(16))) __bf16 v16bf;
typedef __attribute__((ext_vector_type(8)))  __bf16 v8bf;
typedef __attribute__((ext_vector_type(8)))  float  v8f;
typedef __attribute__((ext_vector_type(8)))  unsigned v8u;

// float -> bf16 bits with round-to-nearest-even (no __bf16 arithmetic needed)
__device__ inline unsigned bf16_bits(float x) {
  unsigned u = __float_as_uint(x);
  return (u + 0x7FFFu + ((u >> 16) & 1u)) >> 16;
}
__device__ inline unsigned pack2bf(float lo, float hi) {
  return bf16_bits(lo) | (bf16_bits(hi) << 16);
}

// A-fragment (16x32 bf16): lane m=row; halves 0..15 = K[16*hi .. 16*hi+15]
// (contiguous order, matching load_bfrag)
__device__ inline v16bf load_afrag(const unsigned short* row, int hi) {
  v8bf a0 = *(const v8bf*)(row + 16 * hi);
  v8bf a1 = *(const v8bf*)(row + 16 * hi + 8);
  return __builtin_shufflevector(a0, a1, 0,1,2,3,4,5,6,7,8,9,10,11,12,13,14,15);
}
// B-fragment (32x16 bf16): lane n=col; halves 0..15 = K[16*hi .. 16*hi+15]
// caller passes row pointer already offset by 16*hi
__device__ inline v16bf load_bfrag(const unsigned short* p) {
  v8bf b0 = *(const v8bf*)(p);
  v8bf b1 = *(const v8bf*)(p + 8);
  return __builtin_shufflevector(b0, b1, 0,1,2,3,4,5,6,7,8,9,10,11,12,13,14,15);
}

__device__ inline v8f wmma_bf16(v16bf a, v16bf b, v8f c) {
  return __builtin_amdgcn_wmma_f32_16x16x32_bf16(false, a, false, b, (short)0, c,
                                                 false, false);
}

// ---------------------------------------------------------------------------
// CDNA5 async global->LDS copy (ASYNCcnt path, ISA 08 §4).
// A generic pointer derived from a __shared__ array carries the LDS byte
// address in its low 32 bits (ISA §10.2), so the
// truncated pointer is exactly the VDST operand the instruction wants.
// ---------------------------------------------------------------------------
__device__ inline unsigned lds_off_u32(const void* p) {
  return (unsigned)(unsigned long long)p;
}
__device__ inline void async_ld_b128(const unsigned short* g, unsigned lds) {
  asm volatile("global_load_async_to_lds_b128 %0, %1, off"
               :: "v"(lds), "v"(g) : "memory");
}
__device__ inline void wait_async_le(int pending4) {
  if (pending4)
    asm volatile("s_wait_asynccnt 0x4" ::: "memory");
  else
    asm volatile("s_wait_asynccnt 0x0" ::: "memory");
}

// ---------------------------------------------------------------------------
// f32 -> bf16 conversion (4-wide)
// ---------------------------------------------------------------------------
__global__ __launch_bounds__(256) void cvt_f32_bf16(const float* __restrict__ s,
                                                    unsigned short* __restrict__ d,
                                                    int n4) {
  int i = blockIdx.x * blockDim.x + threadIdx.x;
  if (i >= n4) return;
  float4 f = ((const float4*)s)[i];
  ushort4 r;
  r.x = (unsigned short)bf16_bits(f.x);
  r.y = (unsigned short)bf16_bits(f.y);
  r.z = (unsigned short)bf16_bits(f.z);
  r.w = (unsigned short)bf16_bits(f.w);
  ((ushort4*)d)[i] = r;
}

// ---------------------------------------------------------------------------
// QKV projection:  y = x @ W^T + b, written as bf16 into [B,H,T,HD]
// block = 256 threads (8 waves), tile = 128M x 128N, K-panel = 32,
// double-buffered async LDS pipeline (4 async b128 per thread per panel).
// ---------------------------------------------------------------------------
__global__ __launch_bounds__(256) void gemm_qkv(
    const unsigned short* __restrict__ X,                         // [NTOK, D] bf16
    const unsigned short* __restrict__ Wq,
    const unsigned short* __restrict__ Wk,
    const unsigned short* __restrict__ Wv,
    const float* __restrict__ bq, const float* __restrict__ bk,
    const float* __restrict__ bv,
    unsigned short* __restrict__ Q, unsigned short* __restrict__ K,
    unsigned short* __restrict__ V) {
  const int z = blockIdx.z;
  const unsigned short* W = (z == 0) ? Wq : (z == 1) ? Wk : Wv;
  const float* bias        = (z == 0) ? bq : (z == 1) ? bk : bv;
  unsigned short* Out      = (z == 0) ? Q  : (z == 1) ? K  : V;

  __shared__ unsigned short Xs[2][128 * 32];
  __shared__ unsigned short Ws[2][128 * 32];

  const int tid  = threadIdx.x;
  const int lane = tid & 31, wid = tid >> 5;
  const int hi = lane >> 4, l15 = lane & 15;
  const int m0 = blockIdx.x * 128, n0 = blockIdx.y * 128;
  const int wm = (wid >> 2) * 64, wn = (wid & 3) * 32;

  v8f acc[4][2] = {};

  const int ldrow  = tid >> 1;
  const int ldhalf = (tid & 1) * 16;

  auto issue_panel = [&](int buf, int k0) {
    const unsigned short* gx = X + (m0 + ldrow) * Dn + k0 + ldhalf;
    unsigned lx = lds_off_u32(Xs[buf] + ldrow * 32 + ldhalf);
    async_ld_b128(gx,     lx);
    async_ld_b128(gx + 8, lx + 16);
    const unsigned short* gw = W + (n0 + ldrow) * Dn + k0 + ldhalf;
    unsigned lw = lds_off_u32(Ws[buf] + ldrow * 32 + ldhalf);
    async_ld_b128(gw,     lw);
    async_ld_b128(gw + 8, lw + 16);
  };

  issue_panel(0, 0);
  for (int k0 = 0; k0 < Dn; k0 += 32) {
    const int cur = (k0 >> 5) & 1;
    const int more = (k0 + 32 < Dn);
    if (more) issue_panel(cur ^ 1, k0 + 32);
    wait_async_le(more);          // current panel landed (async done in-order)
    __syncthreads();

    v16bf bfr[2];
#pragma unroll
    for (int nt = 0; nt < 2; ++nt)
      bfr[nt] = load_bfrag(Ws[cur] + (wn + nt * 16 + l15) * 32 + 16 * hi);
#pragma unroll
    for (int mt = 0; mt < 4; ++mt) {
      v16bf afr = load_afrag(Xs[cur] + (wm + mt * 16 + l15) * 32, hi);
      acc[mt][0] = wmma_bf16(afr, bfr[0], acc[mt][0]);
      acc[mt][1] = wmma_bf16(afr, bfr[1], acc[mt][1]);
    }
    __syncthreads();              // done reading buf before it is refilled
  }

  // epilogue: +bias, bf16, scatter into [B,H,T,HD]
#pragma unroll
  for (int nt = 0; nt < 2; ++nt) {
    const int n  = n0 + wn + nt * 16 + l15;   // output feature
    const float bb = bias[n];
    const int h = n >> 6, hd = n & (HDn - 1);
#pragma unroll
    for (int mt = 0; mt < 4; ++mt) {
#pragma unroll
      for (int r = 0; r < 8; ++r) {
        const int m = m0 + wm + mt * 16 + r + 8 * hi;   // token index
        const int b = m >> 11, t = m & (Tn - 1);
        Out[(((b * Hn + h) * Tn) + t) * HDn + hd] =
            (unsigned short)bf16_bits(acc[mt][nt][r] + bb);
      }
    }
  }
}

// ---------------------------------------------------------------------------
// Flash attention over [B,H,T,HD] bf16 -> attn-out bf16 [NTOK, D]
// block = 128 threads (4 waves); each wave owns 16 query rows.
// S^T = K·Q^T (per-lane query columns), online softmax, O^T = V^T·P^T.
// K panel fetched via async-to-LDS; V transposed through registers.
// ---------------------------------------------------------------------------
__global__ __launch_bounds__(128) void flash_attn(
    const unsigned short* __restrict__ Q, const unsigned short* __restrict__ K,
    const unsigned short* __restrict__ V, const float* __restrict__ x_std,
    unsigned short* __restrict__ Out) {
  const int qblk = blockIdx.x, h = blockIdx.y, b = blockIdx.z;
  const int tid = threadIdx.x, lane = tid & 31, wid = tid >> 5;
  const int hi = lane >> 4, l15 = lane & 15;

  __shared__ unsigned short Ks[32 * 64];   // [kv][d]
  __shared__ unsigned short Vt[64 * 32];   // [d][kv]

  const int base = ((b * Hn + h) * Tn) * HDn;
  const unsigned short* qp = Q + base;
  const unsigned short* kp = K + base;
  const unsigned short* vp = V + base;

  const int q0 = qblk * 64 + wid * 16;
  const float sc = 0.125f / x_std[b];      // HD^-0.5 / (tau * sigma), delta = 0

  // Q B-fragments (held for whole kernel): lane = query, halves = head-dim
  v16bf qf[2];
#pragma unroll
  for (int s = 0; s < 2; ++s)
    qf[s] = load_bfrag(qp + (q0 + l15) * HDn + 32 * s + 16 * hi);

  v8f o[4] = {};
  float m_run = -1e30f, l_run = 0.0f;

  // 128 threads x 16 halves = 2048 = 32x64 panel
  const int kvr = (tid * 16) >> 6;     // kv row 0..31
  const int kvc = (tid * 16) & 63;     // d col  0..63 step 16

  for (int kv0 = 0; kv0 < Tn; kv0 += 32) {
    // K panel: async global->LDS
    {
      const unsigned short* gk = kp + (kv0 + kvr) * HDn + kvc;
      unsigned lk = lds_off_u32(Ks + kvr * 64 + kvc);
      async_ld_b128(gk,     lk);
      async_ld_b128(gk + 8, lk + 16);
    }
    // V panel: register transpose into LDS
    {
      const unsigned short* gv = vp + (kv0 + kvr) * HDn + kvc;
      v8bf v0 = *(const v8bf*)(gv);
      v8bf v1 = *(const v8bf*)(gv + 8);
#pragma unroll
      for (int i = 0; i < 8; ++i) {
        Vt[(kvc + i) * 32 + kvr]     = ((const unsigned short*)&v0)[i];
        Vt[(kvc + 8 + i) * 32 + kvr] = ((const unsigned short*)&v1)[i];
      }
    }
    wait_async_le(0);
    __syncthreads();

    // S^T tiles: s0 = kv 0..15, s1 = kv 16..31 (lane = query column)
    v8f s0 = {}, s1 = {};
#pragma unroll
    for (int s = 0; s < 2; ++s) {
      v16bf a0 = load_afrag(Ks + l15 * 64 + 32 * s, hi);
      v16bf a1 = load_afrag(Ks + (16 + l15) * 64 + 32 * s, hi);
      s0 = wmma_bf16(a0, qf[s], s0);
      s1 = wmma_bf16(a1, qf[s], s1);
    }

    // online softmax (per-lane: fixed query, 16 kv values; fold lane-halves)
    float p[16];
    float mx = -1e30f;
#pragma unroll
    for (int i = 0; i < 8; ++i) { p[i] = s0[i] * sc; mx = fmaxf(mx, p[i]); }
#pragma unroll
    for (int i = 0; i < 8; ++i) { p[8 + i] = s1[i] * sc; mx = fmaxf(mx, p[8 + i]); }
    mx = fmaxf(mx, __shfl_xor(mx, 16, 32));
    const float m_new = fmaxf(m_run, mx);
    const float corr  = __expf(m_run - m_new);
    float lsum = 0.0f;
#pragma unroll
    for (int i = 0; i < 16; ++i) { p[i] = __expf(p[i] - m_new); lsum += p[i]; }
    lsum += __shfl_xor(lsum, 16, 32);
    l_run = l_run * corr + lsum;
    m_run = m_new;
#pragma unroll
    for (int dt = 0; dt < 4; ++dt)
#pragma unroll
      for (int r = 0; r < 8; ++r) o[dt][r] *= corr;

    // Assemble P^T as a 32x16 B-fragment (lane = query).
    // lane<16 holds kv {0..7,16..23}; lane>=16 holds kv {8..15,24..31}.
    unsigned pk[8];
#pragma unroll
    for (int i = 0; i < 4; ++i) pk[i]     = pack2bf(p[2 * i], p[2 * i + 1]);
#pragma unroll
    for (int i = 0; i < 4; ++i) pk[4 + i] = pack2bf(p[8 + 2 * i], p[9 + 2 * i]);
    v8u fu;
#pragma unroll
    for (int i = 0; i < 4; ++i) {
      unsigned send = hi ? pk[i] : pk[4 + i];
      unsigned recv = (unsigned)__shfl_xor((int)send, 16, 32);
      fu[i]     = hi ? recv      : pk[i];
      fu[4 + i] = hi ? pk[4 + i] : recv;
    }
    v16bf pf = __builtin_bit_cast(v16bf, fu);

    // O^T tiles: A = V^T rows (head-dim), B = P^T
#pragma unroll
    for (int dt = 0; dt < 4; ++dt) {
      v16bf vfr = load_afrag(Vt + (dt * 16 + l15) * 32, hi);
      o[dt] = wmma_bf16(vfr, pf, o[dt]);
    }
    __syncthreads();
  }

  // normalize + store bf16 attn-out [token, D] (D index = h*64 + d)
  const float inv_l = 1.0f / l_run;
  const int qrow = q0 + l15;
  unsigned short* op = Out + (b * Tn + qrow) * Dn + h * HDn;
#pragma unroll
  for (int dt = 0; dt < 4; ++dt)
#pragma unroll
    for (int r = 0; r < 8; ++r) {
      const int d = dt * 16 + r + 8 * hi;
      op[d] = (unsigned short)bf16_bits(o[dt][r] * inv_l);
    }
}

// ---------------------------------------------------------------------------
// Output projection: out = aout @ Wo^T + bo  (f32 result), async-pipelined
// ---------------------------------------------------------------------------
__global__ __launch_bounds__(256) void gemm_o(
    const unsigned short* __restrict__ A,     // [NTOK, D] bf16
    const unsigned short* __restrict__ Wo,    // [D, D] bf16
    const float* __restrict__ bo, float* __restrict__ Out) {
  __shared__ unsigned short Xs[2][128 * 32];
  __shared__ unsigned short Ws[2][128 * 32];

  const int tid  = threadIdx.x;
  const int lane = tid & 31, wid = tid >> 5;
  const int hi = lane >> 4, l15 = lane & 15;
  const int m0 = blockIdx.x * 128, n0 = blockIdx.y * 128;
  const int wm = (wid >> 2) * 64, wn = (wid & 3) * 32;

  v8f acc[4][2] = {};
  const int ldrow  = tid >> 1;
  const int ldhalf = (tid & 1) * 16;

  auto issue_panel = [&](int buf, int k0) {
    const unsigned short* gx = A + (m0 + ldrow) * Dn + k0 + ldhalf;
    unsigned lx = lds_off_u32(Xs[buf] + ldrow * 32 + ldhalf);
    async_ld_b128(gx,     lx);
    async_ld_b128(gx + 8, lx + 16);
    const unsigned short* gw = Wo + (n0 + ldrow) * Dn + k0 + ldhalf;
    unsigned lw = lds_off_u32(Ws[buf] + ldrow * 32 + ldhalf);
    async_ld_b128(gw,     lw);
    async_ld_b128(gw + 8, lw + 16);
  };

  issue_panel(0, 0);
  for (int k0 = 0; k0 < Dn; k0 += 32) {
    const int cur = (k0 >> 5) & 1;
    const int more = (k0 + 32 < Dn);
    if (more) issue_panel(cur ^ 1, k0 + 32);
    wait_async_le(more);
    __syncthreads();

    v16bf bfr[2];
#pragma unroll
    for (int nt = 0; nt < 2; ++nt)
      bfr[nt] = load_bfrag(Ws[cur] + (wn + nt * 16 + l15) * 32 + 16 * hi);
#pragma unroll
    for (int mt = 0; mt < 4; ++mt) {
      v16bf afr = load_afrag(Xs[cur] + (wm + mt * 16 + l15) * 32, hi);
      acc[mt][0] = wmma_bf16(afr, bfr[0], acc[mt][0]);
      acc[mt][1] = wmma_bf16(afr, bfr[1], acc[mt][1]);
    }
    __syncthreads();
  }

#pragma unroll
  for (int nt = 0; nt < 2; ++nt) {
    const int n  = n0 + wn + nt * 16 + l15;
    const float bb = bo[n];
#pragma unroll
    for (int mt = 0; mt < 4; ++mt) {
#pragma unroll
      for (int r = 0; r < 8; ++r) {
        const int m = m0 + wm + mt * 16 + r + 8 * hi;
        Out[m * Dn + n] = acc[mt][nt][r] + bb;
      }
    }
  }
}

// ---------------------------------------------------------------------------
// Host launcher
// ---------------------------------------------------------------------------
extern "C" void kernel_launch(void* const* d_in, const int* in_sizes, int n_in,
                              void* d_out, int out_size, void* d_ws,
                              size_t ws_size, hipStream_t stream) {
  (void)in_sizes; (void)n_in; (void)out_size; (void)ws_size;
  const float* x     = (const float*)d_in[0];
  const float* x_std = (const float*)d_in[2];
  const float* Wq = (const float*)d_in[3];  const float* bq = (const float*)d_in[4];
  const float* Wk = (const float*)d_in[5];  const float* bk = (const float*)d_in[6];
  const float* Wv = (const float*)d_in[7];  const float* bv = (const float*)d_in[8];
  const float* Wo = (const float*)d_in[9];  const float* bo = (const float*)d_in[10];
  float* out = (float*)d_out;

  // workspace layout (bf16 buffers, all 16B-aligned)
  char* ws = (char*)d_ws;
  const size_t SZ_X = (size_t)NTOK * Dn * 2;      // 16 MB
  const size_t SZ_W = (size_t)Dn * Dn * 2;        //  2 MB
  unsigned short* xb  = (unsigned short*)(ws);
  unsigned short* wqb = (unsigned short*)(ws + SZ_X);
  unsigned short* wkb = (unsigned short*)(ws + SZ_X + SZ_W);
  unsigned short* wvb = (unsigned short*)(ws + SZ_X + 2 * SZ_W);
  unsigned short* wob = (unsigned short*)(ws + SZ_X + 3 * SZ_W);
  unsigned short* qb  = (unsigned short*)(ws + SZ_X + 4 * SZ_W);
  unsigned short* kb  = (unsigned short*)(ws + 2 * SZ_X + 4 * SZ_W);
  unsigned short* vb  = (unsigned short*)(ws + 3 * SZ_X + 4 * SZ_W);
  unsigned short* ab  = (unsigned short*)(ws + 4 * SZ_X + 4 * SZ_W);

  // 1) f32 -> bf16 conversions
  {
    const int n4x = NTOK * Dn / 4;
    cvt_f32_bf16<<<(n4x + 255) / 256, 256, 0, stream>>>(x, xb, n4x);
    const int n4w = Dn * Dn / 4;
    cvt_f32_bf16<<<(n4w + 255) / 256, 256, 0, stream>>>(Wq, wqb, n4w);
    cvt_f32_bf16<<<(n4w + 255) / 256, 256, 0, stream>>>(Wk, wkb, n4w);
    cvt_f32_bf16<<<(n4w + 255) / 256, 256, 0, stream>>>(Wv, wvb, n4w);
    cvt_f32_bf16<<<(n4w + 255) / 256, 256, 0, stream>>>(Wo, wob, n4w);
  }

  // 2) fused QKV projections (z selects proj)
  gemm_qkv<<<dim3(NTOK / 128, Dn / 128, 3), 256, 0, stream>>>(
      xb, wqb, wkb, wvb, bq, bk, bv, qb, kb, vb);

  // 3) flash attention
  flash_attn<<<dim3(Tn / 64, Hn, Bn), 128, 0, stream>>>(qb, kb, vb, x_std, ab);

  // 4) output projection (f32 out)
  gemm_o<<<dim3(NTOK / 128, Dn / 128), 256, 0, stream>>>(ab, wob, bo, out);
}
